// MPNNConv_15006615733821
// MI455X (gfx1250) — matrix operand order repaired and checked by reading, not compiled, not run on hardware
//
#include <hip/hip_runtime.h>

// ---------------------------------------------------------------------------
// MPNNConv for MI455X (gfx1250): bf16 WMMA edge-MLP + atomic scatter-add.
// h (51.2MB) fits in 192MB L2 -> gathers are L2 hits; scatter atomics dominate.
// GEMMs run on v_wmma_f32_16x16x32_bf16 (fp32 accumulate).
// ---------------------------------------------------------------------------

typedef __attribute__((ext_vector_type(16))) __bf16 v16bf;
typedef __attribute__((ext_vector_type(8)))  float  v8f;

#define CIN    128
#define CEDGE  32
#define KTOT   288            // 2*CIN + CEDGE
#define NCH1   9              // KTOT / 32
#define NCH2   4              // CIN  / 32
#define NT     8              // 128 / 16 column tiles (one per wave)
#define TILE_E 16
#define STRA   592            // LDS row stride (bytes) for edge_input: 288*2 + 16 pad
#define STRM   272            // LDS row stride (bytes) for intermediate: 128*2 + 16 pad
#define W1_DW  (NCH1 * NT * 32 * 8)   // 18432 dwords of packed bf16 We1
#define W2_DW  (NCH2 * NT * 32 * 8)   //  8192 dwords of packed bf16 We2

__device__ __forceinline__ unsigned short f2bf(float f) {
  unsigned int u = __float_as_uint(f);
  u += 0x7FFFu + ((u >> 16) & 1u);      // round-to-nearest-even
  return (unsigned short)(u >> 16);
}

union Frag16 { v16bf v; uint4 u[2]; };

// Pre-swizzle weights into per-lane WMMA B-fragment order:
// [kchunk][ntile][lane(32)][vgpr(8 dwords)]; lane&15 = column N, lane>>4 picks
// K half (0-15 / 16-31 of the chunk), dword v packs (K=2v, K=2v+1) as bf16x2.
__global__ void pack_weights_kernel(const float* __restrict__ W1,
                                    const float* __restrict__ W2,
                                    unsigned int* __restrict__ out) {
  int d = blockIdx.x * blockDim.x + threadIdx.x;
  if (d >= W1_DW + W2_DW) return;
  const float* W;
  int dd, base;
  if (d < W1_DW) { W = W1; dd = d;         base = 0;     }
  else           { W = W2; dd = d - W1_DW; base = W1_DW; }
  int v = dd & 7;
  int l = (dd >> 3) & 31;
  int t = (dd >> 8) & 7;
  int c = dd >> 11;
  int kk  = ((l >> 4) << 4) + (v << 1);
  int row = (c << 5) + kk;
  int col = (t << 4) + (l & 15);
  unsigned int lo = f2bf(W[row       * CIN + col]);
  unsigned int hi = f2bf(W[(row + 1) * CIN + col]);
  out[base + dd] = lo | (hi << 16);
}

__global__ __launch_bounds__(256) void mpnn_edge_kernel(
    const float* __restrict__ hin,  float* __restrict__ hout,
    const int*   __restrict__ ei,   const int* __restrict__ ej,
    const float* __restrict__ eattr,
    const unsigned int* __restrict__ wpack,
    const float* __restrict__ be1,  const float* __restrict__ be2,
    int E)
{
  __shared__ uint4 ldsA4[TILE_E * (STRA / 16)];   // edge_input [16 x 288] bf16
  __shared__ uint4 ldsM4[TILE_E * (STRM / 16)];   // relu(x@We1+be1) [16 x 128] bf16
  __shared__ int   sh_ej[TILE_E];
  unsigned short* ldsA = (unsigned short*)ldsA4;
  unsigned short* ldsM = (unsigned short*)ldsM4;

  const int tid   = threadIdx.x;
  const int ebase = blockIdx.x * TILE_E;

  // ---- gather: h_i || h_j || edge_attr  ->  bf16 LDS rows ----
  {
    int row = tid >> 4;
    int c16 = tid & 15;
    int edge = ebase + row;
    int ecl  = edge < E ? edge : (E - 1);
    int si = ei[ecl];
    int sj = ej[ecl];
    if (c16 == 0) sh_ej[row] = sj;
    unsigned short* rp = ldsA + row * (STRA / 2);
    for (int c = c16; c < KTOT; c += 16) {
      float x;
      if      (c < CIN)     x = hin[si * CIN + c];
      else if (c < 2 * CIN) x = hin[sj * CIN + (c - CIN)];
      else                  x = eattr[ecl * CEDGE + (c - 2 * CIN)];
      rp[c] = f2bf(x);
    }
  }
  __syncthreads();

  const int wave = tid >> 5;         // output column tile 0..7
  const int lane = tid & 31;
  const int half = lane >> 4;
  const int n    = lane & 15;        // A-row (M) / C-column within tile
  const int col  = wave * 16 + n;

  // ---- GEMM1: [16x288]_bf16 x [288x128]_bf16 -> f32, K chunks of 32 ----
  v8f acc = {};
  const uint4* w1 = (const uint4*)wpack;
  for (int c = 0; c < NCH1; ++c) {
    Frag16 a, b;
    // A 16x32 layout (matches packed B): lane<16 -> K[0..15], lane>=16 -> K[16..31]
    const unsigned char* p =
        (const unsigned char*)ldsA + n * STRA + c * 64 + half * 32;
    a.u[0] = *(const uint4*)(p);
    a.u[1] = *(const uint4*)(p + 16);
    const uint4* q = w1 + (size_t)((c * NT + wave) * 32 + lane) * 2;
    b.u[0] = q[0];
    b.u[1] = q[1];
    acc = __builtin_amdgcn_wmma_f32_16x16x32_bf16(
        false, a.v, false, b.v, (short)0, acc, false, false);
  }

  // ---- relu(acc + be1) -> bf16 intermediate in LDS ----
  {
    float b1 = be1[col];
    for (int r = 0; r < 8; ++r) {
      float x = acc[r] + b1;                 // C element (M = r + 8*half, N = n)
      x = x > 0.f ? x : 0.f;
      ldsM[(r + half * 8) * (STRM / 2) + col] = f2bf(x);
    }
  }
  __syncthreads();

  // ---- GEMM2: [16x128]_bf16 x [128x128]_bf16 -> f32 ----
  v8f acc2 = {};
  const uint4* w2 = (const uint4*)(wpack + W1_DW);
  for (int c = 0; c < NCH2; ++c) {
    Frag16 a, b;
    const unsigned char* p =
        (const unsigned char*)ldsM + n * STRM + c * 64 + half * 32;
    a.u[0] = *(const uint4*)(p);
    a.u[1] = *(const uint4*)(p + 16);
    const uint4* q = w2 + (size_t)((c * NT + wave) * 32 + lane) * 2;
    b.u[0] = q[0];
    b.u[1] = q[1];
    acc2 = __builtin_amdgcn_wmma_f32_16x16x32_bf16(
        false, a.v, false, b.v, (short)0, acc2, false, false);
  }

  // ---- scatter-add m_ij into hout (segment_sum by ej) ----
  {
    float b2 = be2[col];
    for (int r = 0; r < 8; ++r) {
      int m = r + half * 8;
      if (ebase + m < E) {
        atomicAdd(&hout[(size_t)sh_ej[m] * CIN + col], acc2[r] + b2);
      }
    }
  }
}

__global__ void add_bias_kernel(float* __restrict__ out,
                                const float* __restrict__ bias, int total) {
  int i = blockIdx.x * blockDim.x + threadIdx.x;
  if (i < total) out[i] += bias[i & (CIN - 1)];
}

extern "C" void kernel_launch(void* const* d_in, const int* in_sizes, int n_in,
                              void* d_out, int out_size, void* d_ws, size_t ws_size,
                              hipStream_t stream) {
  const float* x     = (const float*)d_in[0];
  const int*   eidx  = (const int*)d_in[1];   // (2,E) int32
  const float* eattr = (const float*)d_in[2];
  const float* We1   = (const float*)d_in[3];
  const float* be1   = (const float*)d_in[4];
  const float* We2   = (const float*)d_in[5];
  const float* be2   = (const float*)d_in[6];
  const float* bias  = (const float*)d_in[7];

  const int E = in_sizes[1] / 2;
  const int N = in_sizes[0] / CIN;
  const int* ei = eidx;
  const int* ej = eidx + E;

  unsigned int* wpack = (unsigned int*)d_ws;
  float* hA   = (float*)((unsigned char*)d_ws + (size_t)(W1_DW + W2_DW) * 4);
  float* hOut = (float*)d_out;
  size_t hbytes = (size_t)N * CIN * sizeof(float);

  // 1) pack weights into bf16 WMMA B-fragment layout
  {
    int tot = W1_DW + W2_DW;
    pack_weights_kernel<<<(tot + 255) / 256, 256, 0, stream>>>(We1, We2, wpack);
  }

  // 2) h0 = x in both read buffer (ws) and accumulation buffer (d_out)
  hipMemcpyAsync(hA,   x, hbytes, hipMemcpyDeviceToDevice, stream);
  hipMemcpyAsync(hOut, x, hbytes, hipMemcpyDeviceToDevice, stream);

  const int nblk = (E + TILE_E - 1) / TILE_E;

  // 3) step 1: read h0 (hA), accumulate m into hOut -> h1
  mpnn_edge_kernel<<<nblk, 256, 0, stream>>>(hA, hOut, ei, ej, eattr,
                                             wpack, be1, be2, E);

  // 4) step 2: hA = h1; read h1, accumulate into hOut -> h2
  hipMemcpyAsync(hA, hOut, hbytes, hipMemcpyDeviceToDevice, stream);
  mpnn_edge_kernel<<<nblk, 256, 0, stream>>>(hA, hOut, ei, ej, eattr,
                                             wpack, be1, be2, E);

  // 5) out = h2 + bias
  int total = N * CIN;
  add_bias_kernel<<<(total + 255) / 256, 256, 0, stream>>>(hOut, bias, total);
}
